// TSumTDisc_14181982011740
// MI455X (gfx1250) — matrix operand order repaired and checked by reading, not compiled, not run on hardware
//
#include <hip/hip_runtime.h>
#include <cstdint>

#define N_NODES 1048576
#define N_GRAPHS 4096
#define D 64
#define H 40
#define NL 4
#define NG 5
#define WAVES 8                      // 8 wave32 per block, 16-row tile per wave
#define TILE_NODES (WAVES * 16)      // 128 nodes per block
#define NODE_BLOCKS (N_NODES / TILE_NODES)

typedef __bf16 bf16_t;
typedef __attribute__((ext_vector_type(16))) __bf16 v16bf;
typedef __attribute__((ext_vector_type(8)))  float  v8f;

union afrag { v16bf v; bf16_t h[16]; };
union cfrag { v8f   v; float  f[8];  };

__device__ __forceinline__ float lrelu(float v) { return fmaxf(v, 0.01f * v); }

__device__ __forceinline__ v8f wmma_bf16(v16bf a, v16bf b, v8f c) {
    // D = A(16x32 bf16) * B(32x16 bf16) + C(16x16 f32)
    return __builtin_amdgcn_wmma_f32_16x16x32_bf16(false, a, false, b, (short)0, c, false, false);
}

// ---- B-fragment pre-swizzle: w[K x Nw] (fp32, row-major) -> LDS bf16 fragments
// frag index f = t*nKC + kc ; within frag: lane, element e
// element (lane,e) holds  k = 32*kc + 8*(lane>>4) + e + 8*(e>>3) ,  n = 16*t + (lane&15)
__device__ __forceinline__ void fill_bfrags(bf16_t* dst, const float* __restrict__ w,
                                            int K, int Nw, int nT, int nKC,
                                            int tid, int nthr) {
    int total = nT * nKC * 512;
    for (int idx = tid; idx < total; idx += nthr) {
        int f    = idx >> 9;
        int lane = (idx >> 4) & 31;
        int e    = idx & 15;
        int t    = f / nKC;
        int kc   = f - t * nKC;
        int k    = 32 * kc + ((lane >> 4) << 3) + e + ((e >> 3) << 3);
        int n    = 16 * t + (lane & 15);
        float v  = (k < K && n < Nw) ? w[k * Nw + n] : 0.f;
        dst[idx] = (bf16_t)v;
    }
}

// ---- A-fragment (16x32 bf16) element mapping:  k = kbase + e + 8*((e>>3) + (lane>>4))
__device__ __forceinline__ v16bf a_frag_guarded(const float* __restrict__ row,
                                                int kbase, int Kvalid, int lane) {
    afrag a;
    int kh = (lane >> 4) << 3;
#pragma unroll
    for (int e = 0; e < 16; ++e) {
        int k = kbase + e + ((e >> 3) << 3) + kh;
        a.h[e] = (bf16_t)((k < Kvalid) ? row[k] : 0.f);
    }
    return a.v;
}

// Fast path for a full 64-wide fp32 row: K-runs are 8 contiguous floats -> float4 loads.
__device__ __forceinline__ v16bf a_frag64(const float* __restrict__ row, int kc, int lane) {
    int base = 32 * kc + ((lane >> 4) << 3);
    float4 q0 = *(const float4*)(row + base);
    float4 q1 = *(const float4*)(row + base + 4);
    float4 q2 = *(const float4*)(row + base + 16);
    float4 q3 = *(const float4*)(row + base + 20);
    afrag a;
    a.h[0]=(bf16_t)q0.x; a.h[1]=(bf16_t)q0.y; a.h[2]=(bf16_t)q0.z; a.h[3]=(bf16_t)q0.w;
    a.h[4]=(bf16_t)q1.x; a.h[5]=(bf16_t)q1.y; a.h[6]=(bf16_t)q1.z; a.h[7]=(bf16_t)q1.w;
    a.h[8]=(bf16_t)q2.x; a.h[9]=(bf16_t)q2.y; a.h[10]=(bf16_t)q2.z; a.h[11]=(bf16_t)q2.w;
    a.h[12]=(bf16_t)q3.x; a.h[13]=(bf16_t)q3.y; a.h[14]=(bf16_t)q3.z; a.h[15]=(bf16_t)q3.w;
    return a.v;
}

// ============ e-pipeline: e = lrelu(lrelu(src @ ew1) @ ew2);  e->global, segment-sum->aggr
// srcRow: 64 fp32 for row (lane&15) of this wave's 16-row tile (global x or LDS xnew).
__device__ __forceinline__ void e_pipeline(const float* __restrict__ srcRow,
                                           const v16bf (*sBe)[32],          // 6 frags [t*2+kc]
                                           const float* __restrict__ sW2,   // ew2 [40*4]
                                           float (*sHw)[44],                // wave staging [16][44]
                                           int lane, long row0,
                                           const int* __restrict__ batch,
                                           float* __restrict__ e_out,
                                           float* __restrict__ aggr) {
    v16bf a0 = a_frag64(srcRow, 0, lane);
    v16bf a1 = a_frag64(srcRow, 1, lane);
    int mb = (lane >> 4) << 3;
    int nl = lane & 15;
#pragma unroll
    for (int t = 0; t < 3; ++t) {                      // H=40 padded to 48 (3 N-tiles)
        cfrag c{};
        c.v = wmma_bf16(a0, sBe[t * 2 + 0][lane], c.v);
        c.v = wmma_bf16(a1, sBe[t * 2 + 1][lane], c.v);
        int n = 16 * t + nl;
        if (n < H) {
#pragma unroll
            for (int r = 0; r < 8; ++r) sHw[mb + r][n] = lrelu(c.f[r]);
        }
    }
    __syncthreads();
    // tail: [16x40] @ [40x4] — lane owns (row = lane&15, cols j0..j0+1)
    int m  = lane & 15;
    int j0 = (lane >> 4) << 1;
    float acc0 = 0.f, acc1 = 0.f;
    const float* hr = sHw[m];
#pragma unroll
    for (int k = 0; k < H; ++k) {
        float hv = hr[k];
        acc0 = fmaf(hv, sW2[k * NL + j0], acc0);
        acc1 = fmaf(hv, sW2[k * NL + j0 + 1], acc1);
    }
    acc0 = lrelu(acc0); acc1 = lrelu(acc1);
    long node = row0 + m;
    *(float2*)(e_out + node * NL + j0) = make_float2(acc0, acc1);

    int b  = batch[node];
    int b0 = __shfl(b, 0, 32);
    if (__all(b == b0)) {                              // sorted batch: common case
        float s0 = acc0, s1 = acc1;
#pragma unroll
        for (int msk = 1; msk <= 8; msk <<= 1) {       // reduce the 16 rows in each half-wave
            s0 += __shfl_xor(s0, msk, 32);
            s1 += __shfl_xor(s1, msk, 32);
        }
        if (m == 0) {                                  // lanes 0 and 16
            atomicAdd(aggr + (long)b0 * NL + j0,     s0);
            atomicAdd(aggr + (long)b0 * NL + j0 + 1, s1);
        }
    } else {
        atomicAdd(aggr + (long)b * NL + j0,     acc0);
        atomicAdd(aggr + (long)b * NL + j0 + 1, acc1);
    }
}

// ============ o-pipeline: cat=[e, g[batch]]; h=lrelu(cat@ow1); o=h@ow2; sO = 2*x + o
// XSRC_LDS=false: x from global.  XSRC_LDS=true: x read in-place from sO (recomputed x1).
template <bool XSRC_LDS>
__device__ __forceinline__ void o_pipeline(int lane, long row0,
                                           const float* __restrict__ x_glb,
                                           const float* __restrict__ e_src,
                                           const float* __restrict__ g_src,
                                           const int* __restrict__ batch,
                                           const v16bf (*sB1)[32],          // 3 frags (K=9 pad 32)
                                           const v16bf (*sB2)[32],          // 8 frags [t*2+kc]
                                           float (*sCatw)[12], float (*sHw)[44],
                                           float (*sOw)[64]) {
    int m = lane & 15;
    long node = row0 + m;
    if (lane < 16) {
        float4 ev = *(const float4*)(e_src + node * NL);
        sCatw[m][0] = ev.x; sCatw[m][1] = ev.y; sCatw[m][2] = ev.z; sCatw[m][3] = ev.w;
    } else {
        const float* gr = g_src + (long)batch[node] * NG;
        sCatw[m][4] = gr[0]; sCatw[m][5] = gr[1]; sCatw[m][6] = gr[2];
        sCatw[m][7] = gr[3]; sCatw[m][8] = gr[4];
        sCatw[m][9] = 0.f; sCatw[m][10] = 0.f; sCatw[m][11] = 0.f;
    }
    __syncthreads();

    v16bf ac = a_frag_guarded(sCatw[m], 0, NL + NG, lane);   // K=9, zero-padded to 32
    int mb = (lane >> 4) << 3;
    int nl = lane & 15;
#pragma unroll
    for (int t = 0; t < 3; ++t) {
        cfrag c{};
        c.v = wmma_bf16(ac, sB1[t][lane], c.v);
        int n = 16 * t + nl;
        if (n < H) {
#pragma unroll
            for (int r = 0; r < 8; ++r) sHw[mb + r][n] = lrelu(c.f[r]);
        }
    }
    __syncthreads();

    v16bf ah0 = a_frag_guarded(sHw[m], 0,  H, lane);         // K=40 over two 32-chunks
    v16bf ah1 = a_frag_guarded(sHw[m], 32, H, lane);
#pragma unroll
    for (int t = 0; t < 4; ++t) {                            // D=64 -> 4 N-tiles
        cfrag c{};
        c.v = wmma_bf16(ah0, sB2[t * 2 + 0][lane], c.v);
        c.v = wmma_bf16(ah1, sB2[t * 2 + 1][lane], c.v);
        int n = 16 * t + nl;
#pragma unroll
        for (int r = 0; r < 8; ++r) {
            int mm = mb + r;
            float xv;
            if constexpr (XSRC_LDS) xv = sOw[mm][n];                 // x1 recomputed in LDS
            else                    xv = x_glb[(row0 + mm) * D + n]; // coalesced per 16 lanes
            sOw[mm][n] = 2.f * xv + c.f[r];                          // x + (x_in + ffn(cat))
        }
    }
    __syncthreads();
}

// ============ Kernel 1: e0 = ffn(x); aggr0 = segment_sum(e0) ============
__global__ void __launch_bounds__(256)
k_node_e(const float* __restrict__ x, const int* __restrict__ batch,
         const float* __restrict__ ew1, const float* __restrict__ ew2,
         float* __restrict__ e_out, float* __restrict__ aggr) {
    __shared__ v16bf sBe[6][32];
    __shared__ float sW2[H * NL];
    __shared__ float sH[WAVES][16][44];
    int tid = threadIdx.x;
    fill_bfrags((bf16_t*)sBe, ew1, D, H, 3, 2, tid, 256);
    for (int i = tid; i < H * NL; i += 256) sW2[i] = ew2[i];
    __syncthreads();
    int wave = tid >> 5, lane = tid & 31;
    long row0 = ((long)blockIdx.x * WAVES + wave) * 16;
    e_pipeline(x + (row0 + (lane & 15)) * D, sBe, sW2, sH[wave],
               lane, row0, batch, e_out, aggr);
}

// ============ Kernel 2/4: g = ffn(aggr) per graph ============
__global__ void __launch_bounds__(256)
k_graph_g(const float* __restrict__ aggr, const float* __restrict__ w1,
          const float* __restrict__ w2, float* __restrict__ g_out) {
    __shared__ float s1[NL * H];
    __shared__ float s2[H * NG];
    int tid = threadIdx.x;
    for (int i = tid; i < NL * H; i += 256) s1[i] = w1[i];
    for (int i = tid; i < H * NG; i += 256) s2[i] = w2[i];
    __syncthreads();
    int gi = blockIdx.x * 256 + tid;
    float4 a = *(const float4*)(aggr + (long)gi * NL);
    float acc[NG] = {0.f, 0.f, 0.f, 0.f, 0.f};
#pragma unroll 8
    for (int j = 0; j < H; ++j) {
        float h = a.x * s1[j] + a.y * s1[H + j] + a.z * s1[2 * H + j] + a.w * s1[3 * H + j];
        h = lrelu(h);
#pragma unroll
        for (int c = 0; c < NG; ++c) acc[c] = fmaf(h, s2[j * NG + c], acc[c]);
    }
    float* go = g_out + (long)gi * NG;
#pragma unroll
    for (int c = 0; c < NG; ++c) go[c] = lrelu(acc[c]);
}

// ============ Kernel 3: x1 = 2x + o0 (kept in LDS only) -> e1, aggr1 ============
__global__ void __launch_bounds__(256)
k_node_o_fused_e(const float* __restrict__ x, const int* __restrict__ batch,
                 const float* __restrict__ e0, const float* __restrict__ g0,
                 const float* __restrict__ ow1, const float* __restrict__ ow2,
                 const float* __restrict__ ew1n, const float* __restrict__ ew2n,
                 float* __restrict__ e1_out, float* __restrict__ aggr1) {
    __shared__ v16bf sB1[3][32];
    __shared__ v16bf sB2[8][32];
    __shared__ v16bf sBe[6][32];
    __shared__ float sW2[H * NL];
    __shared__ float sCat[WAVES][16][12];
    __shared__ float sH[WAVES][16][44];
    __shared__ float sO[WAVES][16][64];
    int tid = threadIdx.x;
    fill_bfrags((bf16_t*)sB1, ow1, NL + NG, H, 3, 1, tid, 256);
    fill_bfrags((bf16_t*)sB2, ow2, H, D, 4, 2, tid, 256);
    fill_bfrags((bf16_t*)sBe, ew1n, D, H, 3, 2, tid, 256);
    for (int i = tid; i < H * NL; i += 256) sW2[i] = ew2n[i];
    __syncthreads();
    int wave = tid >> 5, lane = tid & 31;
    long row0 = ((long)blockIdx.x * WAVES + wave) * 16;
    o_pipeline<false>(lane, row0, x, e0, g0, batch, sB1, sB2,
                      sCat[wave], sH[wave], sO[wave]);
    // layer-1 e-FFN straight off the LDS-resident x1 rows (x1 never hits HBM)
    e_pipeline(&sO[wave][lane & 15][0], sBe, sW2, sH[wave],
               lane, row0, batch, e1_out, aggr1);
}

// ============ Kernel 5: recompute x1 in LDS, x2 = 2*x1 + o1, pooled += x2 ============
__global__ void __launch_bounds__(256)
k_node_o2_pool(const float* __restrict__ x, const int* __restrict__ batch,
               const float* __restrict__ e0, const float* __restrict__ g0,
               const float* __restrict__ ow1a, const float* __restrict__ ow2a,
               const float* __restrict__ e1, const float* __restrict__ g1,
               const float* __restrict__ ow1b, const float* __restrict__ ow2b,
               float* __restrict__ pooled) {
    __shared__ v16bf sB1a[3][32];
    __shared__ v16bf sB2a[8][32];
    __shared__ v16bf sB1b[3][32];
    __shared__ v16bf sB2b[8][32];
    __shared__ float sCat[WAVES][16][12];
    __shared__ float sH[WAVES][16][44];
    __shared__ float sO[WAVES][16][64];
    int tid = threadIdx.x;
    fill_bfrags((bf16_t*)sB1a, ow1a, NL + NG, H, 3, 1, tid, 256);
    fill_bfrags((bf16_t*)sB2a, ow2a, H, D, 4, 2, tid, 256);
    fill_bfrags((bf16_t*)sB1b, ow1b, NL + NG, H, 3, 1, tid, 256);
    fill_bfrags((bf16_t*)sB2b, ow2b, H, D, 4, 2, tid, 256);
    __syncthreads();
    int wave = tid >> 5, lane = tid & 31;
    long row0 = ((long)blockIdx.x * WAVES + wave) * 16;
    // recompute x1 into sO (cheaper than a 256MB round-trip through HBM)
    o_pipeline<false>(lane, row0, x, e0, g0, batch, sB1a, sB2a,
                      sCat[wave], sH[wave], sO[wave]);
    // layer 1: x2 = 2*x1 + o1, in place in sO
    o_pipeline<true>(lane, row0, x, e1, g1, batch, sB1b, sB2b,
                     sCat[wave], sH[wave], sO[wave]);
    // pooled segment-sum of x2
    int row = lane >> 1, half = lane & 1;
    long node = row0 + row;
    const float* orow = &sO[wave][row][half * 32];
    float v[32];
#pragma unroll
    for (int i = 0; i < 32; i += 4) {
        float4 q = *(const float4*)(orow + i);
        v[i] = q.x; v[i + 1] = q.y; v[i + 2] = q.z; v[i + 3] = q.w;
    }
    int b  = batch[node];
    int b0 = __shfl(b, 0, 32);
    if (__all(b == b0)) {
#pragma unroll
        for (int i = 0; i < 32; ++i) {
#pragma unroll
            for (int msk = 2; msk <= 16; msk <<= 1) v[i] += __shfl_xor(v[i], msk, 32);
        }
        if (row == 0) {                                // lanes 0,1 -> halves 0,1
            float* dst = pooled + (long)b0 * D + half * 32;
#pragma unroll
            for (int i = 0; i < 32; ++i) atomicAdd(dst + i, v[i]);
        }
    } else {
        float* dst = pooled + (long)b * D + half * 32;
#pragma unroll
        for (int i = 0; i < 32; ++i) atomicAdd(dst + i, v[i]);
    }
}

// ============ Kernel 6: discriminator out = lrelu(pooled @ w1) @ w2 ============
__global__ void __launch_bounds__(256)
k_disc(const float* __restrict__ pooled, const float* __restrict__ w1,
       const float* __restrict__ w2, float* __restrict__ out) {
    __shared__ float s1[D * H];
    __shared__ float s2[H];
    int tid = threadIdx.x;
    for (int i = tid; i < D * H; i += 256) s1[i] = w1[i];
    if (tid < H) s2[tid] = w2[tid];
    __syncthreads();
    int g = blockIdx.x * 256 + tid;
    const float* pr = pooled + (long)g * D;
    float row[D];
#pragma unroll
    for (int i = 0; i < D; i += 4) {
        float4 q = *(const float4*)(pr + i);
        row[i] = q.x; row[i + 1] = q.y; row[i + 2] = q.z; row[i + 3] = q.w;
    }
    float acc = 0.f;
    for (int j = 0; j < H; ++j) {
        float h = 0.f;
#pragma unroll
        for (int k = 0; k < D; ++k) h = fmaf(row[k], s1[k * H + j], h);
        acc = fmaf(lrelu(h), s2[j], acc);
    }
    out[g] = acc;
}

extern "C" void kernel_launch(void* const* d_in, const int* in_sizes, int n_in,
                              void* d_out, int out_size, void* d_ws, size_t ws_size,
                              hipStream_t stream) {
    (void)in_sizes; (void)n_in; (void)out_size; (void)ws_size;
    const float* x      = (const float*)d_in[0];
    const int*   batch  = (const int*)d_in[1];
    const float* ew1_0  = (const float*)d_in[2];
    const float* ew2_0  = (const float*)d_in[3];
    const float* gw1_0  = (const float*)d_in[4];
    const float* gw2_0  = (const float*)d_in[5];
    const float* ow1_0  = (const float*)d_in[6];
    const float* ow2_0  = (const float*)d_in[7];
    const float* ew1_1  = (const float*)d_in[8];
    const float* ew2_1  = (const float*)d_in[9];
    const float* gw1_1  = (const float*)d_in[10];
    const float* gw2_1  = (const float*)d_in[11];
    const float* ow1_1  = (const float*)d_in[12];
    const float* ow2_1  = (const float*)d_in[13];
    const float* dw1    = (const float*)d_in[14];
    const float* dw2    = (const float*)d_in[15];
    float* out = (float*)d_out;

    char* w = (char*)d_ws;
    float* e0     = (float*)w; w += (size_t)N_NODES * NL * sizeof(float);   // 16 MB
    float* e1     = (float*)w; w += (size_t)N_NODES * NL * sizeof(float);   // 16 MB
    float* aggr0  = (float*)w; w += (size_t)N_GRAPHS * NL * sizeof(float);
    float* aggr1  = (float*)w; w += (size_t)N_GRAPHS * NL * sizeof(float);
    float* g0     = (float*)w; w += (size_t)N_GRAPHS * NG * sizeof(float);
    float* g1     = (float*)w; w += (size_t)N_GRAPHS * NG * sizeof(float);
    float* pooled = (float*)w; w += (size_t)N_GRAPHS * D  * sizeof(float);  // 1 MB

    hipMemsetAsync(aggr0,  0, (size_t)N_GRAPHS * NL * sizeof(float), stream);
    hipMemsetAsync(aggr1,  0, (size_t)N_GRAPHS * NL * sizeof(float), stream);
    hipMemsetAsync(pooled, 0, (size_t)N_GRAPHS * D  * sizeof(float), stream);

    k_node_e<<<NODE_BLOCKS, 256, 0, stream>>>(x, batch, ew1_0, ew2_0, e0, aggr0);
    k_graph_g<<<N_GRAPHS / 256, 256, 0, stream>>>(aggr0, gw1_0, gw2_0, g0);
    k_node_o_fused_e<<<NODE_BLOCKS, 256, 0, stream>>>(x, batch, e0, g0, ow1_0, ow2_0,
                                                      ew1_1, ew2_1, e1, aggr1);
    k_graph_g<<<N_GRAPHS / 256, 256, 0, stream>>>(aggr1, gw1_1, gw2_1, g1);
    k_node_o2_pool<<<NODE_BLOCKS, 256, 0, stream>>>(x, batch, e0, g0, ow1_0, ow2_0,
                                                    e1, g1, ow1_1, ow2_1, pooled);
    k_disc<<<N_GRAPHS / 256, 256, 0, stream>>>(pooled, dw1, dw2, out);
}
